// GPT2_43782896615794
// MI455X (gfx1250) — matrix operand order repaired and checked by reading, not compiled, not run on hardware
//
#include <hip/hip_runtime.h>
#include <stdint.h>

#define L_ 12
#define E_ 768
#define H_ 12
#define D_ 64
#define V_ 50257
#define T_ 1024
#define B_ 2
#define M_ (B_*T_)   /* 2048 rows */

#if __has_builtin(__builtin_amdgcn_global_load_async_to_lds_b128)
#define USE_ASYNC_LDS 1
#else
#define USE_ASYNC_LDS 0
#endif

typedef __bf16 bf16;
typedef bf16  v8bf  __attribute__((ext_vector_type(8)));
typedef bf16  v16bf __attribute__((ext_vector_type(16)));
typedef float v8f   __attribute__((ext_vector_type(8)));
typedef int   v4i   __attribute__((vector_size(16)));   // async-copy payload type

static __device__ __forceinline__ unsigned short f2bfu(float f) {
    unsigned u = __float_as_uint(f);
    unsigned r = u + 0x7FFFu + ((u >> 16) & 1u);   // round-to-nearest-even
    return (unsigned short)(r >> 16);
}

static __device__ __forceinline__ v8f zero8() {
    v8f z;
#pragma unroll
    for (int i = 0; i < 8; i++) z[i] = 0.0f;
    return z;
}

static __device__ __forceinline__ v16bf mk16(v8bf lo, v8bf hi) {
    v16bf r;
#pragma unroll
    for (int i = 0; i < 8; i++) { r[i] = lo[i]; r[i + 8] = hi[i]; }
    return r;
}

static __device__ __forceinline__ v8f wmma_bf16(v16bf a, v16bf b, v8f c) {
    // D = A(16x32 bf16) x B(32x16 bf16) + C(16x16 f32)
    return __builtin_amdgcn_wmma_f32_16x16x32_bf16(
        false, a, false, b, (short)0, c, false, false);
}

static __device__ __forceinline__ void wait_async0() {
#if USE_ASYNC_LDS
#if __has_builtin(__builtin_amdgcn_s_wait_asynccnt)
    __builtin_amdgcn_s_wait_asynccnt(0);
#else
    asm volatile("s_wait_asynccnt 0" ::: "memory");
#endif
#endif
}

// ---------------------------------------------------------------------------
// Embedding: x[b,t,:] = wte[idx[b,t],:] + wpe[t,:]
// ---------------------------------------------------------------------------
__global__ __launch_bounds__(256) void k_embed(const int* __restrict__ idx,
                                               const float* __restrict__ wte,
                                               const float* __restrict__ wpe,
                                               float* __restrict__ x) {
    int i = blockIdx.x * 256 + threadIdx.x;
    if (i >= M_ * E_) return;
    int r = i / E_, c = i - r * E_;
    int t = r % T_;
    x[i] = wte[(size_t)idx[r] * E_ + c] + wpe[(size_t)t * E_ + c];
}

// ---------------------------------------------------------------------------
// LayerNorm (f32 in -> bf16 out), one block (256 thr) per row, E = 768 = 3*256
// ---------------------------------------------------------------------------
__global__ __launch_bounds__(256) void k_ln(const float* __restrict__ x,
                                            const float* __restrict__ w,
                                            const float* __restrict__ b,
                                            unsigned short* __restrict__ out) {
    const int r   = blockIdx.x;
    const int tid = threadIdx.x;
    const int lane = tid & 31, wid = tid >> 5;
    const float* xr = x + (size_t)r * E_;
    float v0 = xr[tid], v1 = xr[tid + 256], v2 = xr[tid + 512];
    __shared__ float red[8];
    __shared__ float stat[2];

    float s = v0 + v1 + v2;
#pragma unroll
    for (int o = 16; o > 0; o >>= 1) s += __shfl_xor(s, o, 32);
    if (lane == 0) red[wid] = s;
    __syncthreads();
    if (tid == 0) {
        float t = 0.f;
        for (int i = 0; i < 8; i++) t += red[i];
        stat[0] = t * (1.0f / E_);
    }
    __syncthreads();
    float mu = stat[0];
    float d0 = v0 - mu, d1 = v1 - mu, d2 = v2 - mu;
    float s2 = d0 * d0 + d1 * d1 + d2 * d2;
#pragma unroll
    for (int o = 16; o > 0; o >>= 1) s2 += __shfl_xor(s2, o, 32);
    if (lane == 0) red[wid] = s2;
    __syncthreads();
    if (tid == 0) {
        float t = 0.f;
        for (int i = 0; i < 8; i++) t += red[i];
        stat[1] = rsqrtf(t * (1.0f / E_) + 1e-5f);
    }
    __syncthreads();
    float rs = stat[1];
    size_t base = (size_t)r * E_;
    out[base + tid]       = f2bfu(d0 * rs * w[tid]       + b[tid]);
    out[base + tid + 256] = f2bfu(d1 * rs * w[tid + 256] + b[tid + 256]);
    out[base + tid + 512] = f2bfu(d2 * rs * w[tid + 512] + b[tid + 512]);
}

// ---------------------------------------------------------------------------
// WMMA GEMM: C[M,N] = A[M,K](bf16) * B(K,N f32; BT: B is [N,K] f32)
// Block: 256 thr = 8 waves. Block tile 128(M) x 64(N), K step 64.
// Each wave: 16(M) x 64(N) strip = 4 accumulators, 8 WMMAs per K-step.
// Double-buffered LDS; one barrier per K-step.
//  - A tile staged with GLOBAL_LOAD_ASYNC_TO_LDS_B128 (ASYNCcnt) when the
//    toolchain exposes the builtin (M is always a multiple of 128 here, so
//    the async copy needs no bounds guard); register-staged fallback else.
//  - B tile is f32->bf16 converted through VGPRs; next-next stage is warmed
//    with global_prefetch (__builtin_prefetch).
// EPI: 0 = bf16(acc+bias), 1 = bf16(gelu(acc+bias)),
//      2 = f32(acc+bias+resid), 3 = f32(acc)
// ---------------------------------------------------------------------------
template<bool BT, int EPI>
__global__ __launch_bounds__(256) void k_gemm(const unsigned short* __restrict__ A, int lda,
                                              const float* __restrict__ Bm, int ldb,
                                              const float* bias,
                                              const float* resid, int ldr,
                                              float* outF, unsigned short* outH, int ldc,
                                              int Mdim, int Ndim, int Kdim) {
    constexpr int AP = 72, BP = 72;       // padded row strides (144B, 16B-aligned)
    __shared__ __align__(16) unsigned short As[2][128 * AP];  // [buf][128 m][64 k]
    __shared__ __align__(16) unsigned short Bs[2][64 * BP];   // [buf][64 n][64 k]

    const int tid = threadIdx.x;
    const int m0 = blockIdx.y * 128;
    const int n0 = blockIdx.x * 64;
    const int w = tid >> 5, lane = tid & 31;
    const int hf = lane >> 4, l16 = lane & 15;

    v8f acc[4];
#pragma unroll
    for (int nt = 0; nt < 4; nt++) acc[nt] = zero8();

    // ---- A staging --------------------------------------------------------
#if USE_ASYNC_LDS
    auto stageA_async = [&](int k0, int buf) {
#pragma unroll
        for (int i = 0; i < 4; i++) {
            int f = i * 2048 + tid * 8;          // 16B per lane per copy
            int row = f >> 6, col = f & 63;
            __builtin_amdgcn_global_load_async_to_lds_b128(
                (__attribute__((address_space(1))) v4i*)
                    (A + (size_t)(m0 + row) * lda + k0 + col),
                (__attribute__((address_space(3))) v4i*)
                    (&As[buf][row * AP + col]),
                0, 0);
        }
    };
#else
    auto loadA = [&](int k0, uint4 va[4]) {
#pragma unroll
        for (int i = 0; i < 4; i++) {
            int f = i * 2048 + tid * 8;
            int row = f >> 6, col = f & 63;
            int gm = m0 + row;
            uint4 v; v.x = v.y = v.z = v.w = 0u;
            if (gm < Mdim)
                v = *(const uint4*)(A + (size_t)gm * lda + k0 + col);
            va[i] = v;
        }
    };
    auto storeA = [&](int buf, const uint4 va[4]) {
#pragma unroll
        for (int i = 0; i < 4; i++) {
            int f = i * 2048 + tid * 8;
            int row = f >> 6, col = f & 63;
            *(uint4*)&As[buf][row * AP + col] = va[i];
        }
    };
#endif
    // ---- B staging (f32 -> bf16 through VGPRs) ----------------------------
    auto loadB = [&](int k0, float vb[16]) {
        // prefetch the stage after this one (speculative, safe if OOB)
        {
            int pr = tid >> 2, pc = (tid & 3) * 16;
            const float* pf = BT ? (Bm + (size_t)(n0 + pr) * ldb + (k0 + 64) + pc)
                                 : (Bm + (size_t)(k0 + 64 + pr) * ldb + n0 + pc);
            __builtin_prefetch(pf, 0, 1);
        }
#pragma unroll
        for (int i = 0; i < 16; i++) {
            int f = i * 256 + tid;
            float v = 0.0f;
            if (BT) {                     // B stored [N][K]
                int nn = f >> 6, kk = f & 63;
                if (n0 + nn < Ndim) v = Bm[(size_t)(n0 + nn) * ldb + k0 + kk];
            } else {                      // B stored [K][N]
                int kk = f >> 6, nn = f & 63;
                if (n0 + nn < Ndim) v = Bm[(size_t)(k0 + kk) * ldb + n0 + nn];
            }
            vb[i] = v;
        }
    };
    auto storeB = [&](int buf, const float vb[16]) {
#pragma unroll
        for (int i = 0; i < 16; i++) {
            int f = i * 256 + tid;
            int nn, kk;
            if (BT) { nn = f >> 6; kk = f & 63; }
            else    { kk = f >> 6; nn = f & 63; }
            Bs[buf][nn * BP + kk] = f2bfu(vb[i]);
        }
    };

    // ---- prologue: stage k-block 0 ---------------------------------------
    {
#if USE_ASYNC_LDS
        stageA_async(0, 0);
#else
        uint4 va[4];
        loadA(0, va);
        storeA(0, va);
#endif
        float vb[16];
        loadB(0, vb);
        storeB(0, vb);
    }
    wait_async0();
    __syncthreads();

    const int nk = Kdim >> 6;             // K always a multiple of 64 here
    for (int kb = 0; kb < nk; kb++) {
        const int cur = kb & 1;
        const bool more = (kb + 1) < nk;
        float vb[16];
#if USE_ASYNC_LDS
        if (more) { stageA_async((kb + 1) << 6, cur ^ 1); loadB((kb + 1) << 6, vb); }
#else
        uint4 va[4];
        if (more) { loadA((kb + 1) << 6, va); loadB((kb + 1) << 6, vb); }
#endif

        // ---- 8 WMMAs on the current stage --------------------------------
        const unsigned short* Ab = &As[cur][0];
        const unsigned short* Bb = &Bs[cur][0];
#pragma unroll
        for (int c = 0; c < 2; c++) {
            const unsigned short* ap = Ab + (w * 16 + l16) * AP + c * 32 + hf * 16;
            v16bf af = mk16(*(const v8bf*)ap, *(const v8bf*)(ap + 8));
#pragma unroll
            for (int nt = 0; nt < 4; nt++) {
                const unsigned short* bp = Bb + (nt * 16 + l16) * BP + c * 32 + hf * 16;
                acc[nt] = wmma_bf16(af, mk16(*(const v8bf*)bp, *(const v8bf*)(bp + 8)),
                                    acc[nt]);
            }
        }

        if (more) {
#if !USE_ASYNC_LDS
            storeA(cur ^ 1, va);
#endif
            storeB(cur ^ 1, vb);
        }
        wait_async0();
        __syncthreads();
    }

    // ---- epilogue: lane L, reg r -> M = r + hf*8, N = l16 -----------------
#pragma unroll
    for (int nt = 0; nt < 4; nt++) {
        const int nout = n0 + nt * 16 + l16;
#pragma unroll
        for (int r = 0; r < 8; r++) {
            int mout = m0 + w * 16 + hf * 8 + r;
            if (mout < Mdim && nout < Ndim) {
                float v = acc[nt][r];
                if (EPI != 3) v += bias[nout];
                if (EPI == 1) {
                    float c = v;
                    float t = 0.7978845608028654f * (c + 0.044715f * c * c * c);
                    v = 0.5f * c * (1.0f + tanhf(t));
                }
                if (EPI == 2) v += resid[(size_t)mout * ldr + nout];
                if (EPI == 0 || EPI == 1)
                    outH[(size_t)mout * ldc + nout] = f2bfu(v);
                else
                    outF[(size_t)mout * ldc + nout] = v;
            }
        }
    }
}

// ---------------------------------------------------------------------------
// V transpose: vt[b,h,d,t] = qkv[b,t, 2E + h*64 + d]   (bf16)
// ---------------------------------------------------------------------------
__global__ __launch_bounds__(256) void k_vt(const unsigned short* __restrict__ qkv,
                                            unsigned short* __restrict__ vt) {
    int i = blockIdx.x * 256 + threadIdx.x;
    if (i >= B_ * H_ * D_ * T_) return;
    int t = i % T_;
    int d = (i / T_) % D_;
    int h = (i / (T_ * D_)) % H_;
    int b = i / (T_ * D_ * H_);
    vt[i] = qkv[((size_t)(b * T_ + t)) * (3 * E_) + 2 * E_ + h * D_ + d];
}

// ---------------------------------------------------------------------------
// Flash attention: one wave per 16-query tile, online softmax, WMMA for
// S = Q*K^T and O = P*V.  grid = (T/128, B*H), block = 256 (8 waves).
// ---------------------------------------------------------------------------
__global__ __launch_bounds__(256) void k_attn(const unsigned short* __restrict__ qkv,
                                              const unsigned short* __restrict__ vt,
                                              unsigned short* __restrict__ out) {
    __shared__ __align__(16) unsigned short Pl[8][16 * 40];  // per-wave P tile
    const int w = threadIdx.x >> 5, lane = threadIdx.x & 31;
    const int hf = lane >> 4, l16 = lane & 15;
    const int qt = blockIdx.x * 8 + w;
    const int bh = blockIdx.y;
    const int b = bh / H_, h = bh % H_;
    const int ld = 3 * E_;
    const unsigned short* qbase = qkv + ((size_t)b * T_) * ld + h * D_;
    const unsigned short* kbase = qbase + E_;
    const unsigned short* vbase = vt + (size_t)bh * D_ * T_;
    unsigned short* P = &Pl[w][0];

    // Q fragments (two K-chunks over head dim 64), A-layout
    v16bf qa[2];
    {
        const int qrow = qt * 16 + l16;
#pragma unroll
        for (int c = 0; c < 2; c++) {
            const unsigned short* base = qbase + (size_t)qrow * ld + c * 32 + hf * 16;
            qa[c] = mk16(*(const v8bf*)base, *(const v8bf*)(base + 8));
        }
    }

    float mrow[8], lrow[8];
    v8f o[4];
#pragma unroll
    for (int r = 0; r < 8; r++) { mrow[r] = -1e30f; lrow[r] = 0.0f; }
#pragma unroll
    for (int dt = 0; dt < 4; dt++) o[dt] = zero8();

    const int nch = (qt * 16 + 15) / 32 + 1;       // causal key-chunk count
    for (int kc = 0; kc < nch; kc++) {
        // ---- S = Q*K^T for two 16-key subtiles ---------------------------
        v8f s0 = zero8(), s1 = zero8();
        {
            const int key0 = kc * 32 + l16;
            const unsigned short* kb0 = kbase + (size_t)key0 * ld;
#pragma unroll
            for (int c = 0; c < 2; c++) {
                const unsigned short* bp = kb0 + c * 32 + hf * 16;
                s0 = wmma_bf16(qa[c], mk16(*(const v8bf*)bp, *(const v8bf*)(bp + 8)), s0);
            }
            const unsigned short* kb1 = kbase + (size_t)(key0 + 16) * ld;
#pragma unroll
            for (int c = 0; c < 2; c++) {
                const unsigned short* bp = kb1 + c * 32 + hf * 16;
                s1 = wmma_bf16(qa[c], mk16(*(const v8bf*)bp, *(const v8bf*)(bp + 8)), s1);
            }
        }
        // ---- scale + causal mask + online softmax update -----------------
#pragma unroll
        for (int r = 0; r < 8; r++) {
            int q  = qt * 16 + hf * 8 + r;
            int c0 = kc * 32 + l16, c1 = c0 + 16;
            float e0 = s0[r] * 0.125f; if (c0 > q) e0 = -1e30f;
            float e1 = s1[r] * 0.125f; if (c1 > q) e1 = -1e30f;
            float mx = fmaxf(e0, e1);
#pragma unroll
            for (int o2 = 8; o2 > 0; o2 >>= 1) mx = fmaxf(mx, __shfl_xor(mx, o2, 16));
            float mnew  = fmaxf(mrow[r], mx);
            float alpha = __expf(mrow[r] - mnew);
            float p0 = __expf(e0 - mnew), p1 = __expf(e1 - mnew);
            float rs = p0 + p1;
#pragma unroll
            for (int o2 = 8; o2 > 0; o2 >>= 1) rs += __shfl_xor(rs, o2, 16);
            lrow[r] = lrow[r] * alpha + rs;
            mrow[r] = mnew;
#pragma unroll
            for (int dt = 0; dt < 4; dt++) o[dt][r] *= alpha;
            P[(hf * 8 + r) * 40 + l16]      = f2bfu(p0);
            P[(hf * 8 + r) * 40 + 16 + l16] = f2bfu(p1);
        }
        asm volatile("s_wait_dscnt 0" ::: "memory");

        // ---- O += P * V (A-frag from LDS P, B-frag from V^T rows) --------
        const unsigned short* pb = P + l16 * 40 + hf * 16;
        v16bf pa = mk16(*(const v8bf*)pb, *(const v8bf*)(pb + 8));
#pragma unroll
        for (int dt = 0; dt < 4; dt++) {
            int d = dt * 16 + l16;
            const unsigned short* bp = vbase + (size_t)d * T_ + kc * 32 + hf * 16;
            o[dt] = wmma_bf16(pa, mk16(*(const v8bf*)bp, *(const v8bf*)(bp + 8)), o[dt]);
        }
    }

    // ---- normalize + write attn_out (bf16, [M, E]) ------------------------
#pragma unroll
    for (int dt = 0; dt < 4; dt++) {
        int col = h * D_ + dt * 16 + l16;
#pragma unroll
        for (int r = 0; r < 8; r++) {
            int q = qt * 16 + hf * 8 + r;
            float v = o[dt][r] / lrow[r];
            out[((size_t)(b * T_ + q)) * E_ + col] = f2bfu(v);
        }
    }
}

// ---------------------------------------------------------------------------
extern "C" void kernel_launch(void* const* d_in, const int* in_sizes, int n_in,
                              void* d_out, int out_size, void* d_ws, size_t ws_size,
                              hipStream_t stream) {
    (void)in_sizes; (void)n_in; (void)out_size; (void)ws_size;
    const int*   idx   = (const int*)  d_in[0];
    const float* wte   = (const float*)d_in[1];
    const float* wpe   = (const float*)d_in[2];
    const float* ln1w  = (const float*)d_in[3];
    const float* ln1b  = (const float*)d_in[4];
    const float* attnw = (const float*)d_in[5];
    const float* attnb = (const float*)d_in[6];
    const float* apw   = (const float*)d_in[7];
    const float* apb   = (const float*)d_in[8];
    const float* ln2w  = (const float*)d_in[9];
    const float* ln2b  = (const float*)d_in[10];
    const float* fcw   = (const float*)d_in[11];
    const float* fcb   = (const float*)d_in[12];
    const float* fpw   = (const float*)d_in[13];
    const float* fpb   = (const float*)d_in[14];
    const float* lnfw  = (const float*)d_in[15];
    const float* lnfb  = (const float*)d_in[16];
    float* logits = (float*)d_out;

    char* p = (char*)d_ws;
    auto take = [&](size_t bytes) {
        void* q = (void*)p;
        p += (bytes + 255) & ~(size_t)255;
        return q;
    };
    float*          x   = (float*)         take((size_t)M_ * E_ * 4);
    unsigned short* hb  = (unsigned short*)take((size_t)M_ * E_ * 2);
    unsigned short* qkv = (unsigned short*)take((size_t)M_ * 3 * E_ * 2);
    unsigned short* vtb = (unsigned short*)take((size_t)M_ * E_ * 2);
    unsigned short* ao  = (unsigned short*)take((size_t)M_ * E_ * 2);
    unsigned short* fo  = (unsigned short*)take((size_t)M_ * 4 * E_ * 2);

    dim3 blk(256);
    k_embed<<<(M_ * E_ + 255) / 256, blk, 0, stream>>>(idx, wte, wpe, x);

    for (int l = 0; l < L_; l++) {
        k_ln<<<M_, blk, 0, stream>>>(x, ln1w + l * E_, ln1b + l * E_, hb);
        // qkv = LN1(x) @ attn_w + attn_b  -> bf16 [M, 3E]
        k_gemm<false, 0><<<dim3((3 * E_) / 64, M_ / 128), blk, 0, stream>>>(
            hb, E_, attnw + (size_t)l * E_ * 3 * E_, 3 * E_,
            attnb + (size_t)l * 3 * E_, nullptr, 0,
            nullptr, qkv, 3 * E_, M_, 3 * E_, E_);
        k_vt<<<(M_ * E_ + 255) / 256, blk, 0, stream>>>(qkv, vtb);
        k_attn<<<dim3(T_ / 128, B_ * H_), blk, 0, stream>>>(qkv, vtb, ao);
        // x += attn_out @ attn_proj_w + attn_proj_b
        k_gemm<false, 2><<<dim3(E_ / 64, M_ / 128), blk, 0, stream>>>(
            ao, E_, apw + (size_t)l * E_ * E_, E_,
            apb + (size_t)l * E_, x, E_,
            x, nullptr, E_, M_, E_, E_);
        k_ln<<<M_, blk, 0, stream>>>(x, ln2w + l * E_, ln2b + l * E_, hb);
        // fc = gelu(LN2(x) @ fc_w + fc_b) -> bf16 [M, 4E]
        k_gemm<false, 1><<<dim3((4 * E_) / 64, M_ / 128), blk, 0, stream>>>(
            hb, E_, fcw + (size_t)l * E_ * 4 * E_, 4 * E_,
            fcb + (size_t)l * 4 * E_, nullptr, 0,
            nullptr, fo, 4 * E_, M_, 4 * E_, E_);
        // x += fc @ fc_proj_w + fc_proj_b
        k_gemm<false, 2><<<dim3(E_ / 64, M_ / 128), blk, 0, stream>>>(
            fo, 4 * E_, fpw + (size_t)l * 4 * E_ * E_, E_,
            fpb + (size_t)l * E_, x, E_,
            x, nullptr, E_, M_, E_, 4 * E_);
    }

    // final LN then tied lm_head: logits = LNf(x) @ wte^T   (NT GEMM, f32 out)
    k_ln<<<M_, blk, 0, stream>>>(x, lnfw, lnfb, hb);
    k_gemm<true, 3><<<dim3((V_ + 63) / 64, M_ / 128), blk, 0, stream>>>(
        hb, E_, wte, E_, nullptr, nullptr, 0,
        logits, nullptr, V_, M_, V_, E_);
}
